// ResidualAttentionBlock_50139448214102
// MI455X (gfx1250) — matrix operand order repaired and checked by reading, not compiled, not run on hardware
//
#include <hip/hip_runtime.h>
#include <hip/hip_bf16.h>
#include <stdint.h>

// ---------------------------------------------------------------------------
// MI455X (gfx1250, wave32) ResidualAttentionBlock.
// All heavy math via v_wmma_f32_16x16x32_f16 (f16 in, f32 accumulate).
// Round 4: flash kernel split so each query tile is served by a *pair* of
// waves (S subtiles split by key, O accumulation split by d) -> per-wave VGPR
// pressure drops from >256 to ~175, eliminating VGPR-MSB thrash and giving
// the scheduler headroom to overlap ds_load_b128 with the WMMA pipe.
// ---------------------------------------------------------------------------

typedef __attribute__((ext_vector_type(16))) _Float16 v16h;
typedef __attribute__((ext_vector_type(8)))  float    v8f;
typedef unsigned int u32;

#define L_DIM   512
#define D_DIM   512
#define DC_DIM  256
#define T_DIM   128
#define FF_DIM  2048
#define NKEYS   (T_DIM * L_DIM + L_DIM)   // 66048
#define KCONV   (3 * L_DIM)               // 1536

union HU { v16h h; u32 u[8]; };
union WU { u32 u; _Float16 h[2]; };

__device__ __forceinline__ v8f vzero8() {
  v8f z;
#pragma unroll
  for (int i = 0; i < 8; ++i) z[i] = 0.0f;
  return z;
}

__device__ __forceinline__ v16h frag_ones() {
  v16h o;
#pragma unroll
  for (int i = 0; i < 16; ++i) o[i] = (_Float16)1.0f;
  return o;
}

// A fragment (16x32 f16): lane lo holds row m=lo. VGPR v -> K = hi*16+2v(+1).
// base = pointer to this lane's row start.
// With 16B-aligned rows both 4-dword groups lower to ds_load_b128.
__device__ __forceinline__ v16h frag_a_row(const _Float16* base, int hi) {
  HU r;
#pragma unroll
  for (int v = 0; v < 8; ++v) {
    int kb = hi * 16 + 2 * v;
    r.u[v] = *(const u32*)(base + kb);
  }
  return r.h;
}

// B fragment (32x16 f16) where this lane's 16 K-values are contiguous halfs.
__device__ __forceinline__ v16h frag_b_contig(const _Float16* base) {
  HU r;
  const u32* s = (const u32*)base;
#pragma unroll
  for (int j = 0; j < 8; ++j) r.u[j] = s[j];
  return r.h;
}

__device__ __forceinline__ v8f wmma_f16(v16h a, v16h b, v8f c) {
  return __builtin_amdgcn_wmma_f32_16x16x32_f16(false, a, false, b, (short)0, c,
                                                false, false);
}

// ---------------------------------------------------------------------------
// LayerNorm: one block (256 threads) per row of 512.
// ---------------------------------------------------------------------------
__global__ __launch_bounds__(256) void ln_kernel(
    const float* __restrict__ x, const float* __restrict__ g,
    const float* __restrict__ b, float* __restrict__ o32,
    _Float16* __restrict__ o16) {
  __shared__ float red[256];
  const int tid = threadIdx.x;
  const int row = blockIdx.x;
  const float* xr = x + (size_t)row * D_DIM;
  float v0 = xr[tid], v1 = xr[tid + 256];
  red[tid] = v0 + v1;
  __syncthreads();
  for (int s = 128; s > 0; s >>= 1) {
    if (tid < s) red[tid] += red[tid + s];
    __syncthreads();
  }
  float mean = red[0] * (1.0f / D_DIM);
  __syncthreads();
  float d0 = v0 - mean, d1 = v1 - mean;
  red[tid] = d0 * d0 + d1 * d1;
  __syncthreads();
  for (int s = 128; s > 0; s >>= 1) {
    if (tid < s) red[tid] += red[tid + s];
    __syncthreads();
  }
  float inv = rsqrtf(red[0] * (1.0f / D_DIM) + 1e-5f);
  float y0 = d0 * inv * g[tid] + b[tid];
  float y1 = d1 * inv * g[tid + 256] + b[tid + 256];
  if (o32) {
    o32[(size_t)row * D_DIM + tid] = y0;
    o32[(size_t)row * D_DIM + tid + 256] = y1;
  }
  if (o16) {
    o16[(size_t)row * D_DIM + tid] = (_Float16)y0;
    o16[(size_t)row * D_DIM + tid + 256] = (_Float16)y1;
  }
}

// ---------------------------------------------------------------------------
// Transposed im2col for the stride-2 conv (GEMM B^T operand, h-major):
//   XT[h][c] = x1[i][2h + kh - 1],  c = kh*512 + i   (zero padded)
// ---------------------------------------------------------------------------
__global__ void im2colT_kernel(const float* __restrict__ x1,
                               _Float16* __restrict__ XT) {
  int i = blockIdx.x * 256 + threadIdx.x;
  if (i >= DC_DIM * KCONV) return;
  int h = i / KCONV, c = i % KCONV;
  int ii = c & (L_DIM - 1), kh = c >> 9;
  int d = 2 * h + kh - 1;
  float v = (d >= 0 && d < D_DIM) ? x1[(size_t)ii * D_DIM + d] : 0.0f;
  XT[i] = (_Float16)v;
}

// Reorder conv weights w[o][i][kh] -> Wt[o][kh*512 + i], f32 -> f16 (A operand)
__global__ void reorder_w_kernel(const float* __restrict__ w,
                                 _Float16* __restrict__ wt) {
  int i = blockIdx.x * 256 + threadIdx.x;
  if (i >= L_DIM * KCONV) return;
  int o = i / KCONV, c = i % KCONV;
  int ii = c & (L_DIM - 1), kh = c >> 9;
  wt[i] = (_Float16)w[(size_t)o * KCONV + ii * 3 + kh];
}

// Tiled transpose + f32->f16 convert: in (RxC) -> out (CxR).  R,C % 32 == 0.
__global__ __launch_bounds__(256) void tcvt16_kernel(const float* __restrict__ in,
                                                     _Float16* __restrict__ out,
                                                     int R, int C) {
  __shared__ float tile[32][33];
  const int tx = threadIdx.x & 31, ty = threadIdx.x >> 5;
  const int r0 = blockIdx.y * 32, c0 = blockIdx.x * 32;
#pragma unroll
  for (int j = 0; j < 32; j += 8)
    tile[ty + j][tx] = in[(size_t)(r0 + ty + j) * C + c0 + tx];
  __syncthreads();
#pragma unroll
  for (int j = 0; j < 32; j += 8)
    out[(size_t)(c0 + ty + j) * R + r0 + tx] = (_Float16)tile[tx][ty + j];
}

// cached_k * mem_mask -> f16 key buffer rows [0, 65536)
__global__ void mask_cvt_kernel(const float* __restrict__ ck,
                                const int* __restrict__ mask,
                                _Float16* __restrict__ kh) {
  int i = blockIdx.x * 256 + threadIdx.x;
  if (i >= T_DIM * L_DIM * DC_DIM) return;
  int t = i >> 17;  // (i / 256) / 512
  kh[i] = mask[t] ? (_Float16)ck[i] : (_Float16)0.0f;
}

// ---------------------------------------------------------------------------
// Generic WMMA GEMM: C(MxN) = A(MxK,f16,row-major) @ BT(NxK,f16,row-major)^T
// Block = 128 threads (4 waves), 64x64 C tile; wave w owns rows w*16..w*16+15.
// M % 64 == 0, N % 64 == 0, K % 32 == 0 by construction.
// ---------------------------------------------------------------------------
template <bool BIAS, bool RES, bool SILU, bool OUT16, bool OUT32>
__global__ __launch_bounds__(128) void gemm_kernel(
    const _Float16* __restrict__ A, const _Float16* __restrict__ BT,
    const float* __restrict__ bias, const float* __restrict__ res,
    float* __restrict__ C32, _Float16* __restrict__ C16, int M, int N, int K) {
  __shared__ __align__(16) _Float16 As[64 * 40];   // 64 rows x 32 k (+pad)
  __shared__ __align__(16) _Float16 BsT[64 * 40];  // 64 cols x 32 k (+pad)
  const int tid = threadIdx.x;
  const int wid = tid >> 5;
  const int lane = tid & 31;
  const int lo = lane & 15, hi = lane >> 4;
  const int m0 = blockIdx.y * 64, n0 = blockIdx.x * 64;

  v8f acc[4];
#pragma unroll
  for (int f = 0; f < 4; ++f) acc[f] = vzero8();

  for (int k0 = 0; k0 < K; k0 += 32) {
    __syncthreads();
#pragma unroll
    for (int i = 0; i < 8; ++i) {  // A tile: 64 rows x 16 dwords
      int idx = tid + i * 128;
      int r = idx >> 4, c2 = idx & 15;
      *(u32*)&As[r * 40 + 2 * c2] =
          *(const u32*)&A[(size_t)(m0 + r) * K + k0 + 2 * c2];
    }
#pragma unroll
    for (int i = 0; i < 8; ++i) {  // B^T tile: 64 rows x 16 dwords
      int idx = tid + i * 128;
      int r = idx >> 4, c2 = idx & 15;
      *(u32*)&BsT[r * 40 + 2 * c2] =
          *(const u32*)&BT[(size_t)(n0 + r) * K + k0 + 2 * c2];
    }
    __syncthreads();

    v16h a = frag_a_row(&As[(wid * 16 + lo) * 40], hi);
    v16h b = frag_b_contig(&BsT[lo * 40 + hi * 16]);  // f = 0
#pragma unroll
    for (int f = 0; f < 4; ++f) {  // prefetch f+1 before the WMMA on f
      v16h bn = frag_b_contig(&BsT[((((f + 1) & 3) * 16) + lo) * 40 + hi * 16]);
      acc[f] = wmma_f16(a, b, acc[f]);
      b = bn;
    }
  }

#pragma unroll
  for (int f = 0; f < 4; ++f) {
#pragma unroll
    for (int v = 0; v < 8; ++v) {
      int m = m0 + wid * 16 + hi * 8 + v;
      int n = n0 + f * 16 + lo;
      float val = acc[f][v];
      if (BIAS) val += bias[n];
      if (SILU) val = val / (1.0f + __expf(-1.702f * val));
      if (RES) val += res[(size_t)m * N + n];
      if (OUT32) C32[(size_t)m * N + n] = val;
      if (OUT16) C16[(size_t)m * N + n] = (_Float16)val;
    }
  }
}

// ---------------------------------------------------------------------------
// Flash-style attention partials (fixed-baseline softmax):
//   P = exp(QK^T * scale - baseline); writes unnormalized O and row sums.
// Grid: (numQtiles/4, nslices). Block = 256 threads = 8 waves = 4 wave-pairs;
// each pair owns one 16-row query tile. Partner p computes S subtile t=p
// (16x16 of the 16x32 logit tile), P is exchanged through LDS, then each
// partner accumulates half the d-range of O (8 C fragments). Per-wave VGPRs
// ~175 (aq 64 + accO 64) so fragments can stay double-buffered in registers.
// Row sums come from one WMMA against an all-ones B matrix (partner 0).
// ---------------------------------------------------------------------------
__global__ __launch_bounds__(256) void flash_kernel(
    const _Float16* __restrict__ Q, const _Float16* __restrict__ K, int Nk,
    int slice_len, float scale, float baseline, float* __restrict__ pO,
    float* __restrict__ pSum) {
  __shared__ __align__(16) _Float16 Ks[32][264];   // key-major (for S=QK^T)
  __shared__ __align__(16) _Float16 KsT[256][40];  // d-major   (for O+=PK)
  __shared__ __align__(16) _Float16 Ps[4][16][40]; // per-pair P tile
  const int tid = threadIdx.x;
  const int wid = tid >> 5;    // 0..7
  const int pair = wid >> 1;   // query tile within block: 0..3
  const int p = wid & 1;       // partner: S subtile t=p, O d-half p
  const int lane = tid & 31;
  const int lo = lane & 15, hi = lane >> 4;
  const int qt = blockIdx.x * 4 + pair;
  const int s = blockIdx.y;
  const int ns = s * slice_len;
  int ne = ns + slice_len;
  if (ne > Nk) ne = Nk;

  // Q fragments for this pair's 16 rows stay in registers (8 d-chunks).
  v16h aq[8];
  {
    const _Float16* qrow = Q + (size_t)(qt * 16 + lo) * DC_DIM;
#pragma unroll
    for (int c = 0; c < 8; ++c) aq[c] = frag_a_row(qrow + c * 32, hi);
  }

  v8f accO[8];  // this partner's d-range: [p*128, p*128+128)
#pragma unroll
  for (int c = 0; c < 8; ++c) accO[c] = vzero8();
  v8f accS = vzero8();  // used by partner 0 only
  const v16h ones = frag_ones();

  for (int n0 = ns; n0 < ne; n0 += 32) {
    __syncthreads();
    {  // stage 32 keys x 256 halfs, dual layout (4096 dwords / 256 threads)
      const u32* Kg = (const u32*)(K + (size_t)n0 * DC_DIM);
#pragma unroll
      for (int i = 0; i < 16; ++i) {
        int idx = tid + i * 256;
        int r = idx >> 7, c = idx & 127;  // key r, dword c (d = 2c, 2c+1)
        WU w;
        w.u = Kg[r * 128 + c];
        *(u32*)&Ks[r][2 * c] = w.u;
        KsT[2 * c][r] = w.h[0];
        KsT[2 * c + 1][r] = w.h[1];
      }
    }
    __syncthreads();

    // Logits for this partner's 16-key subtile (t=p) -> exp -> P to LDS.
    {
      v8f sacc = vzero8();
      v16h b = frag_b_contig(&Ks[p * 16 + lo][hi * 16]);  // c = 0
#pragma unroll
      for (int c = 0; c < 8; ++c) {  // prefetch c+1 before the WMMA on c
        v16h bn =
            frag_b_contig(&Ks[p * 16 + lo][((c + 1) & 7) * 32 + hi * 16]);
        sacc = wmma_f16(aq[c], b, sacc);
        b = bn;
      }
#pragma unroll
      for (int v = 0; v < 8; ++v) {
        float pr = __expf(sacc[v] * scale - baseline);
        Ps[pair][hi * 8 + v][p * 16 + lo] = (_Float16)pr;  // C -> A transpose
      }
    }
    __syncthreads();  // P tile complete & visible to both partners

    v16h ap = frag_a_row(&Ps[pair][lo][0], hi);
    if (p == 0) accS = wmma_f16(ap, ones, accS);  // row sums of P
    v16h b = frag_b_contig(&KsT[p * 128 + lo][hi * 16]);  // c = 0
#pragma unroll
    for (int c = 0; c < 8; ++c) {  // O += P @ K over this partner's d-half
      v16h bn =
          frag_b_contig(&KsT[p * 128 + ((c + 1) & 7) * 16 + lo][hi * 16]);
      accO[c] = wmma_f16(ap, b, accO[c]);
      b = bn;
    }
  }

  float* po = pO + ((size_t)s * L_DIM + qt * 16) * DC_DIM + p * 128;
#pragma unroll
  for (int c = 0; c < 8; ++c)
#pragma unroll
    for (int v = 0; v < 8; ++v)
      po[(size_t)(hi * 8 + v) * DC_DIM + c * 16 + lo] = accO[c][v];
  if (p == 0 && lo == 0) {
#pragma unroll
    for (int v = 0; v < 8; ++v)
      pSum[s * L_DIM + qt * 16 + hi * 8 + v] = accS[v];
  }
}

// Combine slice partials: O = sum_s pO / sum_s pSum (fixed baseline => add).
__global__ void reduce_kernel(const float* __restrict__ pO,
                              const float* __restrict__ pSum, int S,
                              float* __restrict__ O) {
  int idx = blockIdx.x * 256 + threadIdx.x;
  if (idx >= L_DIM * DC_DIM) return;
  int m = idx >> 8;
  float num = 0.0f, den = 0.0f;
  for (int i = 0; i < S; ++i) {
    num += pO[(size_t)i * L_DIM * DC_DIM + idx];
    den += pSum[i * L_DIM + m];
  }
  O[idx] = num / den;
}

// atten = a_k@k + 0.5*a_self@q + 1.5*q   (the two "+q" terms folded in)
__global__ void combine_kernel(const float* __restrict__ Ok,
                               const float* __restrict__ Os,
                               const float* __restrict__ q,
                               _Float16* __restrict__ out) {
  int i = blockIdx.x * 256 + threadIdx.x;
  if (i >= L_DIM * DC_DIM) return;
  out[i] = (_Float16)(Ok[i] + 0.5f * Os[i] + 1.5f * q[i]);
}

// ---------------------------------------------------------------------------
extern "C" void kernel_launch(void* const* d_in, const int* in_sizes, int n_in,
                              void* d_out, int out_size, void* d_ws,
                              size_t ws_size, hipStream_t stream) {
  const float* x        = (const float*)d_in[0];
  const float* cached_k = (const float*)d_in[1];
  const float* ln1_g    = (const float*)d_in[2];
  const float* ln1_b    = (const float*)d_in[3];
  const float* ln2_g    = (const float*)d_in[4];
  const float* ln2_b    = (const float*)d_in[5];
  const float* w_ck     = (const float*)d_in[6];   // w_compress_k  -> q
  const float* w_cc     = (const float*)d_in[7];   // w_compress_cur -> k_cur
  const float* ap_w     = (const float*)d_in[8];
  const float* ap_b     = (const float*)d_in[9];
  const float* fc_w     = (const float*)d_in[10];
  const float* fc_b     = (const float*)d_in[11];
  const float* proj_w   = (const float*)d_in[12];
  const float* proj_b   = (const float*)d_in[13];
  const int*   mem_mask = (const int*)d_in[14];
  float* out = (float*)d_out;

  char* ws = (char*)d_ws;
  size_t off = 0;
  auto carve = [&](size_t bytes) -> void* {
    off = (off + 255) & ~(size_t)255;
    void* p = ws + off;
    off += bytes;
    return p;
  };

  const int NSLICES = 16;
  float*    x1f    = (float*)carve((size_t)L_DIM * D_DIM * 4);
  float*    qf     = (float*)carve((size_t)L_DIM * DC_DIM * 4);
  _Float16* XimT   = (_Float16*)carve((size_t)DC_DIM * KCONV * 2);
  _Float16* Wtq    = (_Float16*)carve((size_t)L_DIM * KCONV * 2);
  _Float16* Wtc    = (_Float16*)carve((size_t)L_DIM * KCONV * 2);
  _Float16* apwT   = (_Float16*)carve((size_t)D_DIM * DC_DIM * 2);
  _Float16* fcwT   = (_Float16*)carve((size_t)FF_DIM * D_DIM * 2);
  _Float16* pjwT   = (_Float16*)carve((size_t)D_DIM * FF_DIM * 2);
  _Float16* qh     = (_Float16*)carve((size_t)L_DIM * DC_DIM * 2);
  _Float16* kh     = (_Float16*)carve((size_t)NKEYS * DC_DIM * 2);
  float*    pO     = (float*)carve((size_t)NSLICES * L_DIM * DC_DIM * 4);
  float*    pSum   = (float*)carve((size_t)NSLICES * L_DIM * 4);
  float*    pOs    = (float*)carve((size_t)L_DIM * DC_DIM * 4);
  float*    pSums  = (float*)carve((size_t)L_DIM * 4);
  float*    Ok     = (float*)carve((size_t)L_DIM * DC_DIM * 4);
  float*    Osf    = (float*)carve((size_t)L_DIM * DC_DIM * 4);
  _Float16* attenC = (_Float16*)carve((size_t)L_DIM * DC_DIM * 2);
  float*    x2     = (float*)carve((size_t)L_DIM * D_DIM * 4);
  _Float16* x2ln   = (_Float16*)carve((size_t)L_DIM * D_DIM * 2);
  _Float16* u16    = (_Float16*)carve((size_t)L_DIM * FF_DIM * 2);
  (void)ws_size; (void)n_in; (void)in_sizes; (void)out_size;

  // 1. x1 = LN1(x)
  ln_kernel<<<L_DIM, 256, 0, stream>>>(x, ln1_g, ln1_b, x1f, (_Float16*)nullptr);

  // 2. Precision staging (A operands row-major, B operands transposed)
  im2colT_kernel<<<(DC_DIM * KCONV) / 256, 256, 0, stream>>>(x1f, XimT);
  reorder_w_kernel<<<(L_DIM * KCONV) / 256, 256, 0, stream>>>(w_ck, Wtq);
  reorder_w_kernel<<<(L_DIM * KCONV) / 256, 256, 0, stream>>>(w_cc, Wtc);
  tcvt16_kernel<<<dim3(D_DIM / 32, DC_DIM / 32), 256, 0, stream>>>(
      ap_w, apwT, DC_DIM, D_DIM);                     // (256x512) -> (512x256)
  tcvt16_kernel<<<dim3(FF_DIM / 32, D_DIM / 32), 256, 0, stream>>>(
      fc_w, fcwT, D_DIM, FF_DIM);                     // (512x2048)->(2048x512)
  tcvt16_kernel<<<dim3(D_DIM / 32, FF_DIM / 32), 256, 0, stream>>>(
      proj_w, pjwT, FF_DIM, D_DIM);                   // (2048x512)->(512x2048)
  mask_cvt_kernel<<<(T_DIM * L_DIM * DC_DIM) / 256, 256, 0, stream>>>(
      cached_k, mem_mask, kh);

  // 3. conv_pool as GEMM: q (f32+f16) and k_cur (f16, appended to key buffer)
  gemm_kernel<false, false, false, true, true>
      <<<dim3(DC_DIM / 64, L_DIM / 64), 128, 0, stream>>>(
          Wtq, XimT, nullptr, nullptr, qf, qh, L_DIM, DC_DIM, KCONV);
  gemm_kernel<false, false, false, true, false>
      <<<dim3(DC_DIM / 64, L_DIM / 64), 128, 0, stream>>>(
          Wtc, XimT, nullptr, nullptr, nullptr,
          kh + (size_t)(T_DIM * L_DIM) * DC_DIM, L_DIM, DC_DIM, KCONV);

  // 4. Attention over 66048 keys (16 key slices) + self-attention (512 keys).
  //    Fixed baselines: long attn logits ~N(0,1) -> 3.0; self-attn diagonal
  //    ~|q|^2/16 ~ 16 -> 16.0 keeps exp() within f16 range.
  const float scale = 1.0f / 16.0f;  // 1/sqrt(DC)
  flash_kernel<<<dim3(L_DIM / 64, NSLICES), 256, 0, stream>>>(
      qh, kh, NKEYS, NKEYS / NSLICES, scale, 3.0f, pO, pSum);
  reduce_kernel<<<(L_DIM * DC_DIM) / 256, 256, 0, stream>>>(pO, pSum, NSLICES,
                                                            Ok);
  flash_kernel<<<dim3(L_DIM / 64, 1), 256, 0, stream>>>(
      qh, qh, L_DIM, L_DIM, scale, 16.0f, pOs, pSums);
  reduce_kernel<<<(L_DIM * DC_DIM) / 256, 256, 0, stream>>>(pOs, pSums, 1, Osf);

  // 5. atten = Ok + 0.5*Oself + 1.5*q ;  x2 = x1 + atten@ap_w + ap_b
  combine_kernel<<<(L_DIM * DC_DIM) / 256, 256, 0, stream>>>(Ok, Osf, qf,
                                                             attenC);
  gemm_kernel<true, true, false, false, true>
      <<<dim3(D_DIM / 64, L_DIM / 64), 128, 0, stream>>>(
          attenC, apwT, ap_b, x1f, x2, nullptr, L_DIM, D_DIM, DC_DIM);

  // 6. MLP: h = LN2(x2); u = silu1702(h@fc_w+fc_b); out = x2 + u@proj_w+proj_b
  ln_kernel<<<L_DIM, 256, 0, stream>>>(x2, ln2_g, ln2_b, (float*)nullptr, x2ln);
  gemm_kernel<true, false, true, true, false>
      <<<dim3(FF_DIM / 64, L_DIM / 64), 128, 0, stream>>>(
          x2ln, fcwT, fc_b, nullptr, nullptr, u16, L_DIM, FF_DIM, D_DIM);
  gemm_kernel<true, true, false, false, true>
      <<<dim3(D_DIM / 64, L_DIM / 64), 128, 0, stream>>>(
          u16, pjwT, proj_b, x2, out, nullptr, L_DIM, D_DIM, FF_DIM);
}
